// GPT_90555090469699
// MI455X (gfx1250) — compile-verified, matrix operand order checked by reading, not run on hardware
//
#include <hip/hip_runtime.h>

// ---------------- model constants ----------------
#define LAYERS 8
#define NH     16
#define NKV    4
#define DMODEL 1024
#define HDIM   64
#define SEQ    1024
#define NVOCAB 32000
#define DFF    4096

typedef __attribute__((ext_vector_type(16))) __bf16 bf16x16;
typedef __attribute__((ext_vector_type(8)))  float  f32x8;
typedef unsigned int u32x4 __attribute__((ext_vector_type(4)));
typedef int          i32x8 __attribute__((ext_vector_type(8)));
typedef int          i32x4 __attribute__((ext_vector_type(4)));

#if __has_builtin(__builtin_amdgcn_tensor_load_to_lds)
#define USE_TDM 1
#else
#define USE_TDM 0
#endif

union FragU { bf16x16 v; uint4 q[2]; };

__device__ __forceinline__ unsigned short f2bf(float f) {
    unsigned int u = __float_as_uint(f);
    unsigned int r = u + 0x7FFFu + ((u >> 16) & 1u);   // round-to-nearest-even
    return (unsigned short)(r >> 16);
}

#if USE_TDM
// LDS byte offset of a __shared__ object (flat LDS aperture truncates to offset)
__device__ __forceinline__ unsigned ldsOff(const void* p) {
    return (unsigned)(unsigned long long)p;
}
// Issue one TDM 2D tile load: rows x rowlen bf16 elements, row stride = strideElems,
// packed contiguously into LDS at ldsByteOff. D# layout per CDNA5 ISA.
// This toolchain's builtin takes 6 args: (g0 u32x4, g1 i32x8, g2 i32x4, g3 i32x4,
// extra i32x8 (zero-filled), cpol i32).
__device__ __forceinline__ void tdm_load_2d(const unsigned short* gsrc, unsigned ldsByteOff,
                                            int rows, int rowlen, int strideElems) {
    unsigned long long ga = (unsigned long long)gsrc;
    u32x4 g0;
    g0.x = 1u;                                             // count=1, user descriptor
    g0.y = ldsByteOff;                                     // lds_addr
    g0.z = (unsigned)(ga & 0xFFFFFFFFu);                   // global_addr[31:0]
    g0.w = (unsigned)((ga >> 32) & 0x01FFFFFFull) | 0x80000000u; // addr[56:32] | type=2
    i32x8 g1;
    g1[0] = 0x10000;                                       // data_size=1 (2 bytes)
    g1[1] = (rowlen & 0xFFFF) << 16;                       // tensor_dim0[15:0]
    g1[2] = (((unsigned)rowlen >> 16) & 0xFFFF) | ((rows & 0xFFFF) << 16); // dim0 hi | dim1 lo
    g1[3] = (((unsigned)rows >> 16) & 0xFFFF) | ((rowlen & 0xFFFF) << 16); // dim1 hi | tile_dim0
    g1[4] = rows & 0xFFFF;                                 // tile_dim1 (tile_dim2=0)
    g1[5] = strideElems;                                   // tensor_dim0_stride[31:0]
    g1[6] = 0;
    g1[7] = 0;
    i32x4 gz4 = {0, 0, 0, 0};
    i32x8 gz8 = {0, 0, 0, 0, 0, 0, 0, 0};
    __builtin_amdgcn_tensor_load_to_lds(g0, g1, gz4, gz4, gz8, 0);
}
#endif

// ---------------- fp32 -> bf16 with transpose: W (K,N) -> Wt (N,K), batched ----------------
__global__ void convT_k(const float* __restrict__ s, unsigned short* __restrict__ d,
                        int K, int N, int total) {
    for (int idx = blockIdx.x * 256 + threadIdx.x; idx < total; idx += gridDim.x * 256) {
        int mat = idx / (K * N);
        int r   = idx - mat * K * N;
        int n = r / K, k = r % K;
        d[idx] = f2bf(s[(size_t)mat * K * N + (size_t)k * N + n]);
    }
}

// ---------------- embedding + bigram hash ----------------
__global__ void embed_k(const int* __restrict__ tok, const float* __restrict__ wte,
                        const float* __restrict__ big, const float* __restrict__ bscale,
                        float* __restrict__ x) {
    int idx = blockIdx.x * 256 + threadIdx.x;
    if (idx >= SEQ * DMODEL) return;
    int s = idx / DMODEL, d = idx % DMODEL;
    int t = tok[s];
    int hidx;
    if (s == 0) hidx = 65535;
    else {
        int a = (int)(36313u * (unsigned)t);
        int b = (int)(27191u * (unsigned)tok[s - 1]);
        int h = a ^ b;
        int r = h % 65535; if (r < 0) r += 65535;   // python-style mod
        hidx = r;
    }
    x[idx] = wte[(size_t)t * DMODEL + d] + big[(size_t)hidx * DMODEL + d] * bscale[0];
}

// ---------------- rope tables: cos/sin (SEQ x 32) ----------------
__global__ void ropetab_k(float* __restrict__ cosT, float* __restrict__ sinT) {
    int idx = blockIdx.x * 256 + threadIdx.x;
    if (idx >= SEQ * 32) return;
    int s = idx / 32, i = idx % 32;
    float invf = expf(-logf(10000.0f) * (float)i / 32.0f);
    float f = (float)s * invf;
    cosT[idx] = cosf(f);
    sinT[idx] = sinf(f);
}

// ---------------- rmsnorm: one block per row ----------------
__global__ void rmsnorm_k(const float* __restrict__ x, float* __restrict__ outF,
                          unsigned short* __restrict__ outB, int cols) {
    const float* r = x + (size_t)blockIdx.x * cols;
    __shared__ float red[256];
    float s = 0.f;
    for (int j = threadIdx.x; j < cols; j += 256) { float v = r[j]; s += v * v; }
    red[threadIdx.x] = s; __syncthreads();
    for (int t = 128; t; t >>= 1) { if (threadIdx.x < t) red[threadIdx.x] += red[threadIdx.x + t]; __syncthreads(); }
    float inv = rsqrtf(red[0] / (float)cols + 1e-6f);
    for (int j = threadIdx.x; j < cols; j += 256) {
        float v = r[j] * inv;
        if (outF) outF[(size_t)blockIdx.x * cols + j] = v;
        outB[(size_t)blockIdx.x * cols + j] = f2bf(v);
    }
}

// ------- per-(s,head) rmsnorm + rope (+ optional gain), one wave32 each -------
__global__ void headpost_k(const float* __restrict__ in, unsigned short* __restrict__ out,
                           const float* __restrict__ cosT, const float* __restrict__ sinT,
                           const float* __restrict__ gain, int nheads) {
    int wid  = (blockIdx.x << 3) + (threadIdx.x >> 5);
    int lane = threadIdx.x & 31;
    int s = wid / nheads, h = wid % nheads;
    const float* base = in + ((size_t)s * nheads + h) * HDIM;
    float a = base[lane], b = base[lane + 32];
    float ss = a * a + b * b;
    #pragma unroll
    for (int o = 16; o; o >>= 1) ss += __shfl_xor(ss, o, 32);
    float inv = rsqrtf(ss / (float)HDIM + 1e-6f);
    a *= inv; b *= inv;
    float c  = cosT[s * 32 + lane];
    float sn = sinT[s * 32 + lane];
    float g  = gain ? gain[h] : 1.0f;
    float o1 = ( a * c + b * sn) * g;
    float o2 = (-a * sn + b * c) * g;
    unsigned short* ob = out + ((size_t)s * nheads + h) * HDIM;
    ob[lane]      = f2bf(o1);
    ob[lane + 32] = f2bf(o2);
}

// ------- value residual mix; writes V^T (KVD x SEQ) bf16 for K-contiguous GEMM B -------
__global__ void vmix_k(const float* __restrict__ vr, float* __restrict__ v0,
                       unsigned short* __restrict__ vbT, const float* __restrict__ lam,
                       int layer, int n) {
    int i = blockIdx.x * 256 + threadIdx.x;
    if (i >= n) return;
    float l0 = lam[0], l1 = lam[1];
    float raw = vr[i];
    float m;
    if (layer == 0) { v0[i] = raw; m = l1 * raw; }
    else            { m = l0 * v0[i] + l1 * raw; }
    int s = i / (NKV * HDIM), c = i % (NKV * HDIM);
    vbT[(size_t)c * SEQ + s] = f2bf(m);
}

// ---------------- causal softmax: one block per (head,row), bf16 out ----------------
__global__ void softmax_k(const float* __restrict__ sc, unsigned short* __restrict__ pb) {
    int row = blockIdx.x % SEQ;
    int h   = blockIdx.x / SEQ;
    const float* r = sc + ((size_t)h * SEQ + row) * SEQ;
    unsigned short* o = pb + ((size_t)h * SEQ + row) * SEQ;
    int valid = row + 1;
    const float scale = 0.125f;   // 1/sqrt(64)
    __shared__ float red[256];
    float mx = -1e30f;
    for (int j = threadIdx.x; j < valid; j += 256) mx = fmaxf(mx, r[j] * scale);
    red[threadIdx.x] = mx; __syncthreads();
    for (int t = 128; t; t >>= 1) { if (threadIdx.x < t) red[threadIdx.x] = fmaxf(red[threadIdx.x], red[threadIdx.x + t]); __syncthreads(); }
    mx = red[0]; __syncthreads();
    float sum = 0.f;
    for (int j = threadIdx.x; j < valid; j += 256) sum += __expf(r[j] * scale - mx);
    red[threadIdx.x] = sum; __syncthreads();
    for (int t = 128; t; t >>= 1) { if (threadIdx.x < t) red[threadIdx.x] += red[threadIdx.x + t]; __syncthreads(); }
    float inv = 1.0f / red[0];
    for (int j = threadIdx.x; j < SEQ; j += 256) {
        float p = (j < valid) ? __expf(r[j] * scale - mx) * inv : 0.0f;
        o[j] = f2bf(p);
    }
}

// ---------------- sigmoid gate + bf16 conversion of y, one wave32 per (s,h) ----------------
__global__ void gate_k(const float* __restrict__ xn, const float* __restrict__ gw,
                       const float* __restrict__ gb, const float* __restrict__ yf,
                       unsigned short* __restrict__ yb) {
    int wid  = (blockIdx.x << 3) + (threadIdx.x >> 5);
    int lane = threadIdx.x & 31;
    int s = wid / NH, h = wid % NH;
    const float* xr = xn + (size_t)s * DMODEL;
    float acc = 0.f;
    for (int d = lane; d < DMODEL; d += 32) acc += xr[d] * gw[d * NH + h];
    #pragma unroll
    for (int o = 16; o; o >>= 1) acc += __shfl_xor(acc, o, 32);
    float g = 1.0f / (1.0f + __expf(-(acc + gb[h])));
    size_t off = ((size_t)s * NH + h) * HDIM;
    yb[off + lane]      = f2bf(yf[off + lane] * g);
    yb[off + lane + 32] = f2bf(yf[off + lane + 32] * g);
}

// ---------------- squared leaky-relu activation -> bf16 ----------------
__global__ void act_k(const float* __restrict__ hf, unsigned short* __restrict__ hb, int n) {
    int i = blockIdx.x * 256 + threadIdx.x;
    if (i >= n) return;
    float u = hf[i];
    float t = (u >= 0.f) ? u : 0.5f * u;
    hb[i] = f2bf(t * t);
}

// ---------------- residual add ----------------
__global__ void add_k(float* __restrict__ x, const float* __restrict__ t, int n) {
    int i = blockIdx.x * 256 + threadIdx.x;
    if (i < n) x[i] += t[i];
}

// =======================================================================
// Batched BF16 GEMM, C(f32) = A(bf16, M x K row-major) @ Bt(bf16, N x K
// row-major)^T.  Block: 256 threads = 8 wave32, tile M=128 x N=64, K-step
// 32, double-buffered LDS.  Wave 0 DMAs both tiles for step k+32 via
// tensor_load_to_lds (TENSORcnt) while all waves run 4x
// v_wmma_f32_16x16x32_bf16 on step k.  Batch z with per-operand element
// strides; B batch index divided by bdiv (GQA head -> kv-head mapping).
// =======================================================================
__global__ void __launch_bounds__(256)
gemm_bf16(const unsigned short* __restrict__ A, const unsigned short* __restrict__ Bt,
          float* __restrict__ C, int K, int lda, int ldb, int ldc,
          size_t aBatch, size_t bBatch, size_t cBatch, int bdiv) {
    __shared__ __align__(128) unsigned short sA[2][128 * 32];
    __shared__ __align__(128) unsigned short sB[2][64 * 32];   // [n][k]

    const int tid  = threadIdx.x;
    const int wave = tid >> 5, lane = tid & 31;
    const int wr = wave >> 1, wc = wave & 1;
    const int z  = blockIdx.z;
    const int m0 = blockIdx.y * 128;
    const int n0 = blockIdx.x * 64;
    const unsigned short* Abase = A  + (size_t)z * aBatch + (size_t)m0 * lda;
    const unsigned short* Bbase = Bt + (size_t)(z / bdiv) * bBatch + (size_t)n0 * ldb;
    float* Cb = C + (size_t)z * cBatch;

    const int lrow  = lane & 15;
    const int kbase = (lane < 16) ? 0 : 8;

    f32x8 acc[2][2];
    f32x8 zero = {0.f, 0.f, 0.f, 0.f, 0.f, 0.f, 0.f, 0.f};
    acc[0][0] = zero; acc[0][1] = zero; acc[1][0] = zero; acc[1][1] = zero;

    // fragment layout: lanes 0-15 hold K{0..7,16..23}, lanes 16-31 K{8..15,24..31}
    auto compute_tile = [&](int b) {
        FragU fa[2], fb[2];
        #pragma unroll
        for (int t = 0; t < 2; t++) {
            int r = wr * 32 + t * 16 + lrow;
            fa[t].q[0] = *(const uint4*)&sA[b][r * 32 + kbase];
            fa[t].q[1] = *(const uint4*)&sA[b][r * 32 + kbase + 16];
            int c = wc * 32 + t * 16 + lrow;
            fb[t].q[0] = *(const uint4*)&sB[b][c * 32 + kbase];
            fb[t].q[1] = *(const uint4*)&sB[b][c * 32 + kbase + 16];
        }
        #pragma unroll
        for (int i = 0; i < 2; i++)
            #pragma unroll
            for (int j = 0; j < 2; j++)
                acc[i][j] = __builtin_amdgcn_wmma_f32_16x16x32_bf16(
                    false, fa[i].v, false, fb[j].v, (short)0, acc[i][j], false, false);
    };

#if USE_TDM
    unsigned offA0 = ldsOff(&sA[0][0]), offA1 = ldsOff(&sA[1][0]);
    unsigned offB0 = ldsOff(&sB[0][0]), offB1 = ldsOff(&sB[1][0]);
    if (wave == 0) {
        tdm_load_2d(Abase, offA0, 128, 32, lda);
        tdm_load_2d(Bbase, offB0, 64, 32, ldb);
        __builtin_amdgcn_s_wait_tensorcnt(0);
    }
    __syncthreads();
    int buf = 0;
    for (int kb = 0; kb < K; kb += 32) {
        int nxt = buf ^ 1;
        if (wave == 0 && kb + 32 < K) {          // prefetch next K-step via TDM
            tdm_load_2d(Abase + kb + 32, nxt ? offA1 : offA0, 128, 32, lda);
            tdm_load_2d(Bbase + kb + 32, nxt ? offB1 : offB0, 64, 32, ldb);
        }
        compute_tile(buf);                        // WMMAs overlap the DMA
        if (wave == 0) __builtin_amdgcn_s_wait_tensorcnt(0);
        __syncthreads();
        buf = nxt;
    }
#else
    for (int kb = 0; kb < K; kb += 32) {
        {
            int row = tid >> 1, part = tid & 1;
            const unsigned short* src = Abase + (size_t)row * lda + kb + part * 16;
            *(uint4*)&sA[0][row * 32 + part * 16]     = *(const uint4*)src;
            *(uint4*)&sA[0][row * 32 + part * 16 + 8] = *(const uint4*)(src + 8);
        }
        {
            int n = tid >> 2, part = tid & 3;
            *(uint4*)&sB[0][n * 32 + part * 8] =
                *(const uint4*)(Bbase + (size_t)n * ldb + kb + part * 8);
        }
        __syncthreads();
        compute_tile(0);
        __syncthreads();
    }
#endif

    // ---- epilogue: VGPR e, lanes 0-15 -> M=e, lanes 16-31 -> M=e+8 ----
    const int roff = (lane >= 16) ? 8 : 0;
    #pragma unroll
    for (int i = 0; i < 2; i++)
        #pragma unroll
        for (int j = 0; j < 2; j++)
            #pragma unroll
            for (int e = 0; e < 8; e++) {
                int gm = m0 + wr * 32 + i * 16 + roff + e;
                int gn = n0 + wc * 32 + j * 16 + lrow;
                Cb[(size_t)gm * ldc + gn] = acc[i][j][e];
            }
}

// =======================================================================
// host side
// =======================================================================
static inline void launch_gemm(const unsigned short* A, const unsigned short* Bt, float* C,
                               int M, int N, int K, int lda, int ldb, int ldc,
                               size_t aB, size_t bB, size_t cB, int bdiv, int batch,
                               hipStream_t st) {
    dim3 grid(N / 64, M / 128, batch);
    gemm_bf16<<<grid, 256, 0, st>>>(A, Bt, C, K, lda, ldb, ldc, aB, bB, cB, bdiv);
}

extern "C" void kernel_launch(void* const* d_in, const int* in_sizes, int n_in,
                              void* d_out, int out_size, void* d_ws, size_t ws_size,
                              hipStream_t stream) {
    (void)in_sizes; (void)n_in; (void)out_size; (void)ws_size;
    const int*   tok     = (const int*)  d_in[0];
    const float* wte     = (const float*)d_in[1];
    const float* big     = (const float*)d_in[2];
    const float* bscale  = (const float*)d_in[3];
    const float* Wq      = (const float*)d_in[4];
    const float* Wk      = (const float*)d_in[5];
    const float* Wv      = (const float*)d_in[6];
    const float* Wo      = (const float*)d_in[7];
    const float* q_gain  = (const float*)d_in[8];
    const float* vr_lam  = (const float*)d_in[9];
    const float* gate_w  = (const float*)d_in[10];
    const float* gate_b  = (const float*)d_in[11];
    const float* mlp_fc  = (const float*)d_in[12];
    const float* mlp_pj  = (const float*)d_in[13];
    const float* lm_head = (const float*)d_in[14];
    float* out = (float*)d_out;

    // ---- workspace carve (bump allocator, ~370 MB total) ----
    char* p = (char*)d_ws;
    auto alloc = [&](size_t bytes) -> void* {
        void* r = (void*)p;
        p += (bytes + 255) & ~(size_t)255;
        return r;
    };
    const int KVD = NKV * HDIM;   // 256
    const size_t nWq = (size_t)LAYERS * DMODEL * DMODEL;
    const size_t nWk = (size_t)LAYERS * DMODEL * KVD;
    const size_t nFc = (size_t)LAYERS * DMODEL * DFF;
    const size_t nLm = (size_t)DMODEL * NVOCAB;

    unsigned short* WqT  = (unsigned short*)alloc(nWq * 2);   // (N,K) bf16 per layer
    unsigned short* WkT  = (unsigned short*)alloc(nWk * 2);
    unsigned short* WvT  = (unsigned short*)alloc(nWk * 2);
    unsigned short* WoT  = (unsigned short*)alloc(nWq * 2);
    unsigned short* WfcT = (unsigned short*)alloc(nFc * 2);
    unsigned short* WpjT = (unsigned short*)alloc(nFc * 2);
    unsigned short* lmT  = (unsigned short*)alloc(nLm * 2);

    float* x      = (float*)alloc((size_t)SEQ * DMODEL * 4);
    float* xn     = (float*)alloc((size_t)SEQ * DMODEL * 4);
    unsigned short* xnb = (unsigned short*)alloc((size_t)SEQ * DMODEL * 2);
    float* qf     = (float*)alloc((size_t)SEQ * DMODEL * 4);
    float* kf     = (float*)alloc((size_t)SEQ * KVD * 4);
    float* vf     = (float*)alloc((size_t)SEQ * KVD * 4);
    float* v0     = (float*)alloc((size_t)SEQ * KVD * 4);
    unsigned short* qb  = (unsigned short*)alloc((size_t)SEQ * DMODEL * 2);
    unsigned short* kb  = (unsigned short*)alloc((size_t)SEQ * KVD * 2);
    unsigned short* vbT = (unsigned short*)alloc((size_t)KVD * SEQ * 2);
    float* scores = (float*)alloc((size_t)NH * SEQ * SEQ * 4);
    unsigned short* pb = (unsigned short*)alloc((size_t)NH * SEQ * SEQ * 2);
    float* yf     = (float*)alloc((size_t)SEQ * DMODEL * 4);
    unsigned short* yb = (unsigned short*)alloc((size_t)SEQ * DMODEL * 2);
    float* tmp    = (float*)alloc((size_t)SEQ * DMODEL * 4);
    float* hf     = (float*)alloc((size_t)SEQ * DFF * 4);
    unsigned short* hb = (unsigned short*)alloc((size_t)SEQ * DFF * 2);
    float* cosT   = (float*)alloc((size_t)SEQ * 32 * 4);
    float* sinT   = (float*)alloc((size_t)SEQ * 32 * 4);

    // ---- convert + transpose all GEMM weights to bf16 (N,K) each call ----
    convT_k<<<4096, 256, 0, stream>>>(Wq,      WqT,  DMODEL, DMODEL, (int)nWq);
    convT_k<<<4096, 256, 0, stream>>>(Wk,      WkT,  DMODEL, KVD,    (int)nWk);
    convT_k<<<4096, 256, 0, stream>>>(Wv,      WvT,  DMODEL, KVD,    (int)nWk);
    convT_k<<<4096, 256, 0, stream>>>(Wo,      WoT,  DMODEL, DMODEL, (int)nWq);
    convT_k<<<4096, 256, 0, stream>>>(mlp_fc,  WfcT, DMODEL, DFF,    (int)nFc);
    convT_k<<<4096, 256, 0, stream>>>(mlp_pj,  WpjT, DFF,    DMODEL, (int)nFc);
    convT_k<<<4096, 256, 0, stream>>>(lm_head, lmT,  DMODEL, NVOCAB, (int)nLm);

    ropetab_k<<<(SEQ * 32 + 255) / 256, 256, 0, stream>>>(cosT, sinT);
    embed_k<<<(SEQ * DMODEL + 255) / 256, 256, 0, stream>>>(tok, wte, big, bscale, x);

    for (int i = 0; i < LAYERS; i++) {
        // pre-attention rmsnorm (fp32 xn kept for the gate)
        rmsnorm_k<<<SEQ, 256, 0, stream>>>(x, xn, xnb, DMODEL);

        // QKV projections (B^T layout, ldb = K)
        launch_gemm(xnb, WqT + (size_t)i * DMODEL * DMODEL, qf, SEQ, DMODEL, DMODEL,
                    DMODEL, DMODEL, DMODEL, 0, 0, 0, 1, 1, stream);
        launch_gemm(xnb, WkT + (size_t)i * KVD * DMODEL, kf, SEQ, KVD, DMODEL,
                    DMODEL, DMODEL, KVD, 0, 0, 0, 1, 1, stream);
        launch_gemm(xnb, WvT + (size_t)i * KVD * DMODEL, vf, SEQ, KVD, DMODEL,
                    DMODEL, DMODEL, KVD, 0, 0, 0, 1, 1, stream);

        // per-head rmsnorm + rope (+ q gain), value residual mix (writes V^T)
        headpost_k<<<SEQ * NH / 8, 256, 0, stream>>>(qf, qb, cosT, sinT, q_gain + i * NH, NH);
        headpost_k<<<SEQ * NKV / 8, 256, 0, stream>>>(kf, kb, cosT, sinT, nullptr, NKV);
        vmix_k<<<(SEQ * KVD + 255) / 256, 256, 0, stream>>>(vf, v0, vbT, vr_lam + i * 2, i,
                                                            SEQ * KVD);

        // attention scores: per head z -> Q_h (S x 64) @ K_kv^T  (K rows of kb are K-contig)
        launch_gemm(qb, kb, scores, SEQ, SEQ, HDIM,
                    DMODEL, KVD, SEQ,
                    (size_t)HDIM, (size_t)HDIM, (size_t)SEQ * SEQ, NH / NKV, NH, stream);
        softmax_k<<<NH * SEQ, 256, 0, stream>>>(scores, pb);
        // context: per head z -> P_h (S x S) @ V_kv  (B = V^T, KVD x SEQ)
        launch_gemm(pb, vbT, yf, SEQ, HDIM, SEQ,
                    SEQ, SEQ, DMODEL,
                    (size_t)SEQ * SEQ, (size_t)HDIM * SEQ, (size_t)HDIM, NH / NKV, NH, stream);

        // sigmoid gate, then output projection + residual
        gate_k<<<SEQ * NH / 8, 256, 0, stream>>>(xn, gate_w + (size_t)i * DMODEL * NH,
                                                 gate_b + i * NH, yf, yb);
        launch_gemm(yb, WoT + (size_t)i * DMODEL * DMODEL, tmp, SEQ, DMODEL, DMODEL,
                    DMODEL, DMODEL, DMODEL, 0, 0, 0, 1, 1, stream);
        add_k<<<(SEQ * DMODEL + 255) / 256, 256, 0, stream>>>(x, tmp, SEQ * DMODEL);

        // MLP
        rmsnorm_k<<<SEQ, 256, 0, stream>>>(x, nullptr, xnb, DMODEL);
        launch_gemm(xnb, WfcT + (size_t)i * DFF * DMODEL, hf, SEQ, DFF, DMODEL,
                    DMODEL, DMODEL, DFF, 0, 0, 0, 1, 1, stream);
        act_k<<<(SEQ * DFF + 255) / 256, 256, 0, stream>>>(hf, hb, SEQ * DFF);
        launch_gemm(hb, WpjT + (size_t)i * DMODEL * DFF, tmp, SEQ, DMODEL, DFF,
                    DFF, DFF, DMODEL, 0, 0, 0, 1, 1, stream);
        add_k<<<(SEQ * DMODEL + 255) / 256, 256, 0, stream>>>(x, tmp, SEQ * DMODEL);
    }

    // final norm + lm head -> d_out (f32, 1024 x 32000)
    rmsnorm_k<<<SEQ, 256, 0, stream>>>(x, nullptr, xnb, DMODEL);
    launch_gemm(xnb, lmT, out, SEQ, NVOCAB, DMODEL,
                DMODEL, DMODEL, NVOCAB, 0, 0, 0, 1, 1, stream);
}
